// gatcase30_80814104641730
// MI455X (gfx1250) — matrix operand order repaired and checked by reading, not compiled, not run on hardware
//
#include <hip/hip_runtime.h>
#include <hip/hip_bf16.h>
#include <math.h>

// ---------------------------------------------------------------------------
// Model constants (from reference)
// ---------------------------------------------------------------------------
#define D_     256
#define H_     4
#define NN_    30
#define B_     64
#define NT_    1920          // B_*NN_
#define E_     55680
#define ET_    57600         // E_ + NT_ (self loops)
#define NVH_   15
#define PAIRS_ 210
#define EROWS_ 13440         // B_*PAIRS_

#define ACT_NONE  0
#define ACT_LEAKY 1
#define ACT_RELU  2
#define ACT_GELU  3

#define CDIV(a,b) (((a)+(b)-1)/(b))

typedef __attribute__((ext_vector_type(16))) _Float16 v16h;
typedef __attribute__((ext_vector_type(8)))  float    v8f;

// ---------------------------------------------------------------------------
// Device helpers
// ---------------------------------------------------------------------------
__device__ __forceinline__ float wave_sum(float v) {
  #pragma unroll
  for (int o = 16; o > 0; o >>= 1) v += __shfl_xor(v, o, 32);
  return v;
}
__device__ __forceinline__ float wave_max(float v) {
  #pragma unroll
  for (int o = 16; o > 0; o >>= 1) v = fmaxf(v, __shfl_xor(v, o, 32));
  return v;
}
// monotone float<->uint encoding for atomicMax on floats
__device__ __forceinline__ unsigned fenc(float f) {
  unsigned u = __float_as_uint(f);
  return (u & 0x80000000u) ? ~u : (u | 0x80000000u);
}
__device__ __forceinline__ float fdec(unsigned e) {
  return (e & 0x80000000u) ? __uint_as_float(e & 0x7FFFFFFFu) : __uint_as_float(~e);
}
__device__ __forceinline__ float apply_act(float v, int act) {
  if (act == ACT_LEAKY) return v > 0.f ? v : 0.01f * v;
  if (act == ACT_RELU)  return v > 0.f ? v : 0.f;
  if (act == ACT_GELU)  return 0.5f * v * (1.f + erff(v * 0.70710678118f));
  return v;
}

// ---------------------------------------------------------------------------
// Generic WMMA GEMM: C[M,N] = act(A[M,K](f16) @ B + bias) (+C if acc)
// B is supplied TRANSPOSED: BT[N,K] f16, so each lane's B fragment
// (column n, 16 consecutive K) is 32 contiguous bytes -> global_load_b128.
// Each wave computes a 16x64 strip: one A fragment feeds 4 independent WMMAs.
// Fragment k order: contiguous, k = 16*(lane/16) + s, same in A and B (wave32).
// ---------------------------------------------------------------------------
__global__ void kgemm(const _Float16* __restrict__ A, const _Float16* __restrict__ BT,
                      const float* __restrict__ bias, float* __restrict__ C,
                      _Float16* __restrict__ C16, int M, int N, int K, int act, int acc)
{
  int wave = threadIdx.x >> 5;
  int lane = threadIdx.x & 31;
  int ntG  = (N >> 4) >> 2;                    // groups of 4 N-tiles
  int tile = blockIdx.x * 8 + wave;
  int ntiles = (M >> 4) * ntG;
  if (tile >= ntiles) return;                  // wave-uniform: EXEC all-ones for WMMA
  int tg = tile % ntG, tm = tile / ntG;
  int tn0 = tg * 4;

  const int m   = lane & 15;                   // A-row within tile for this lane
  const int ka  = (lane >> 4) << 4;            // A K sub-block (0 or 16)
  const int kbb = (lane >> 4) << 4;            // B K sub-block (0 or 16)
  const int n   = lane & 15;
  const int rbase = tm * 16 + ((lane >> 4) << 3);

  v8f c0 = {}, c1 = {}, c2 = {}, c3 = {};
  if (acc) {
    #pragma unroll
    for (int r = 0; r < 8; ++r) {
      size_t ro = (size_t)(rbase + r) * N;
      c0[r] = C[ro + (tn0 + 0) * 16 + n];
      c1[r] = C[ro + (tn0 + 1) * 16 + n];
      c2[r] = C[ro + (tn0 + 2) * 16 + n];
      c3[r] = C[ro + (tn0 + 3) * 16 + n];
    }
  }
  const _Float16* Arow = A  + (size_t)(tm * 16 + m) * K + ka;
  const _Float16* B0   = BT + (size_t)((tn0 + 0) * 16 + n) * K + kbb;
  const _Float16* B1   = BT + (size_t)((tn0 + 1) * 16 + n) * K + kbb;
  const _Float16* B2   = BT + (size_t)((tn0 + 2) * 16 + n) * K + kbb;
  const _Float16* B3   = BT + (size_t)((tn0 + 3) * 16 + n) * K + kbb;

  for (int k0 = 0; k0 < K; k0 += 32) {
    v16h a, b0, b1, b2, b3;
    #pragma unroll
    for (int j = 0; j < 8; ++j) { a[j] = Arow[k0 + j]; a[8 + j] = Arow[k0 + 8 + j]; }
    #pragma unroll
    for (int j = 0; j < 16; ++j) {
      b0[j] = B0[k0 + j]; b1[j] = B1[k0 + j]; b2[j] = B2[k0 + j]; b3[j] = B3[k0 + j];
    }
    c0 = __builtin_amdgcn_wmma_f32_16x16x32_f16(false, a, false, b0, (short)0, c0, false, false);
    c1 = __builtin_amdgcn_wmma_f32_16x16x32_f16(false, a, false, b1, (short)0, c1, false, false);
    c2 = __builtin_amdgcn_wmma_f32_16x16x32_f16(false, a, false, b2, (short)0, c2, false, false);
    c3 = __builtin_amdgcn_wmma_f32_16x16x32_f16(false, a, false, b3, (short)0, c3, false, false);
  }
  #pragma unroll
  for (int t = 0; t < 4; ++t) {
    int col = (tn0 + t) * 16 + n;
    float bv = bias ? bias[col] : 0.f;
    const v8f& cc = (t == 0) ? c0 : (t == 1) ? c1 : (t == 2) ? c2 : c3;
    #pragma unroll
    for (int r = 0; r < 8; ++r) {
      float v = apply_act(cc[r] + bv, act);
      C[(size_t)(rbase + r) * N + col] = v;
      if (C16) C16[(size_t)(rbase + r) * N + col] = (_Float16)v;
    }
  }
}

// ---------------------------------------------------------------------------
// Conv1d(k=7, pad=3) as 7 accumulated WMMA GEMM strips (16x64 per wave).
// Xpad: (B_,36,D_) f16 zero-padded rows. Wc: [7][O=256][I=256] f16
// (pre-transposed so each lane's B fragment is contiguous). Bias+ReLU fused.
// ---------------------------------------------------------------------------
__global__ void kconvgemm(const _Float16* __restrict__ Xpad, const _Float16* __restrict__ Wc,
                          const float* __restrict__ bias, float* __restrict__ C)
{
  int wave = threadIdx.x >> 5;
  int lane = threadIdx.x & 31;
  int tile = blockIdx.x * 8 + wave;
  const int ntG = (D_ / 16) / 4;               // 4
  if (tile >= (NT_ / 16) * ntG) return;
  int tg = tile % ntG, tm = tile / ntG;
  int tn0 = tg * 4;

  const int m   = lane & 15;
  const int ka  = (lane >> 4) << 4;
  const int kbb = (lane >> 4) << 4;
  const int n   = lane & 15;
  const int rbase = tm * 16 + ((lane >> 4) << 3);

  int rowm = tm * 16 + m;                      // output row for A fragment
  int bb = rowm / NN_, tt = rowm % NN_;
  int prow = bb * 36 + tt;                     // padded-row base (t+dk indexes [0,35])

  v8f c0 = {}, c1 = {}, c2 = {}, c3 = {};
  for (int dk = 0; dk < 7; ++dk) {
    const _Float16* Arow = Xpad + (size_t)(prow + dk) * D_ + ka;
    const _Float16* Wb   = Wc + (size_t)dk * D_ * D_ + kbb;
    const _Float16* B0 = Wb + (size_t)((tn0 + 0) * 16 + n) * D_;
    const _Float16* B1 = Wb + (size_t)((tn0 + 1) * 16 + n) * D_;
    const _Float16* B2 = Wb + (size_t)((tn0 + 2) * 16 + n) * D_;
    const _Float16* B3 = Wb + (size_t)((tn0 + 3) * 16 + n) * D_;
    for (int k0 = 0; k0 < D_; k0 += 32) {
      v16h a, b0, b1, b2, b3;
      #pragma unroll
      for (int j = 0; j < 8; ++j) { a[j] = Arow[k0 + j]; a[8 + j] = Arow[k0 + 8 + j]; }
      #pragma unroll
      for (int j = 0; j < 16; ++j) {
        b0[j] = B0[k0 + j]; b1[j] = B1[k0 + j]; b2[j] = B2[k0 + j]; b3[j] = B3[k0 + j];
      }
      c0 = __builtin_amdgcn_wmma_f32_16x16x32_f16(false, a, false, b0, (short)0, c0, false, false);
      c1 = __builtin_amdgcn_wmma_f32_16x16x32_f16(false, a, false, b1, (short)0, c1, false, false);
      c2 = __builtin_amdgcn_wmma_f32_16x16x32_f16(false, a, false, b2, (short)0, c2, false, false);
      c3 = __builtin_amdgcn_wmma_f32_16x16x32_f16(false, a, false, b3, (short)0, c3, false, false);
    }
  }
  #pragma unroll
  for (int t = 0; t < 4; ++t) {
    int col = (tn0 + t) * 16 + n;
    float bv = bias[col];
    const v8f& cc = (t == 0) ? c0 : (t == 1) ? c1 : (t == 2) ? c2 : c3;
    #pragma unroll
    for (int r = 0; r < 8; ++r) {
      float v = cc[r] + bv;
      C[(size_t)(rbase + r) * D_ + col] = v > 0.f ? v : 0.f;  // ReLU
    }
  }
}

// ---------------------------------------------------------------------------
// Elementwise / utility kernels
// ---------------------------------------------------------------------------
__global__ void kzero(float* p, int n) {
  int t = blockIdx.x * 256 + threadIdx.x;
  if (t < n) p[t] = 0.f;
}
__global__ void kcvt(const float* __restrict__ s, _Float16* __restrict__ d, int n) {
  int t = blockIdx.x * 256 + threadIdx.x;
  if (t < n) d[t] = (_Float16)s[t];
}
// transpose-convert: W (K,N) f32 row-major -> WT (N,K) f16
__global__ void kcvtT(const float* __restrict__ w, _Float16* __restrict__ d, int K, int N) {
  int t = blockIdx.x * 256 + threadIdx.x;
  if (t >= K * N) return;
  int nn = t / K, k = t % K;
  d[t] = (_Float16)w[(size_t)k * N + nn];
}
// conv_W (O,I,7) f32 -> Wc16 [dk][O][I] f16 (B-transposed per tap)
__global__ void kcvtconv(const float* __restrict__ w, _Float16* __restrict__ d) {
  int t = blockIdx.x * 256 + threadIdx.x;
  if (t >= 7 * D_ * D_) return;
  int dk = t >> 16;               // /65536
  int r  = t & 65535;
  int o  = r >> 8, i = r & 255;
  d[t] = (_Float16)w[(size_t)(o * D_ + i) * 7 + dk];
}
// build zero-padded (B_,36,D_) f16 from xc (NT_,D_) f32
__global__ void kpad(const float* __restrict__ xc, _Float16* __restrict__ pad) {
  int t = blockIdx.x * 256 + threadIdx.x;
  if (t >= B_ * 36 * D_) return;
  int c = t & 255;
  int q = (t >> 8) % 36;
  int b = (t >> 8) / 36;
  float v = (q >= 3 && q < 33) ? xc[(size_t)(b * NN_ + q - 3) * D_ + c] : 0.f;
  pad[t] = (_Float16)v;
}
__global__ void kadd3(float* __restrict__ g, const float* __restrict__ c,
                      const float* __restrict__ h, int n) {
  int t = blockIdx.x * 256 + threadIdx.x;
  if (t < n) g[t] += c[t] + h[t];
}

// ---------------------------------------------------------------------------
// LayerNorm over width 256, wave per row. Optional f16 mirror.
// ---------------------------------------------------------------------------
__global__ void kln(const float* __restrict__ X, const float* __restrict__ gam,
                    const float* __restrict__ bet, float* __restrict__ Y,
                    _Float16* __restrict__ Y16, int rows)
{
  int wave = threadIdx.x >> 5, lane = threadIdx.x & 31;
  int row = blockIdx.x * 8 + wave;
  if (row >= rows) return;
  const float* xr = X + (size_t)row * D_;
  float v[8];
  float s = 0.f;
  #pragma unroll
  for (int j = 0; j < 8; ++j) { v[j] = xr[lane + j * 32]; s += v[j]; }
  s = wave_sum(s);
  float mean = s * (1.f / D_);
  float q = 0.f;
  #pragma unroll
  for (int j = 0; j < 8; ++j) { float d = v[j] - mean; q += d * d; }
  q = wave_sum(q);
  float rstd = rsqrtf(q * (1.f / D_) + 1e-5f);
  #pragma unroll
  for (int j = 0; j < 8; ++j) {
    int c = lane + j * 32;
    float y = (v[j] - mean) * rstd * gam[c] + bet[c];
    Y[(size_t)row * D_ + c] = y;
    if (Y16) Y16[(size_t)row * D_ + c] = (_Float16)y;
  }
}

// ---------------------------------------------------------------------------
// GATv2 edge kernels (edge list = edge_index ++ self-loops)
// ---------------------------------------------------------------------------
__global__ void kgat_logits(const float* __restrict__ xl, const float* __restrict__ xr,
                            const float* __restrict__ att, const int* __restrict__ eidx,
                            float* __restrict__ logits, unsigned* __restrict__ mbuf)
{
  int wave = threadIdx.x >> 5, lane = threadIdx.x & 31;
  int e = blockIdx.x * 8 + wave;
  if (e >= ET_) return;
  int src = (e < E_) ? eidx[e]      : (e - E_);
  int dst = (e < E_) ? eidx[E_ + e] : (e - E_);
  const float* pl = xl + (size_t)src * (H_ * D_);
  const float* pr = xr + (size_t)dst * (H_ * D_);
  #pragma unroll
  for (int h = 0; h < H_; ++h) {
    float s = 0.f;
    #pragma unroll
    for (int j = 0; j < 8; ++j) {
      int c = h * D_ + lane + j * 32;
      float v = pl[c] + pr[c];
      v = v > 0.f ? v : 0.2f * v;                  // leaky_relu 0.2
      s += v * att[h * D_ + lane + j * 32];
    }
    s = wave_sum(s);
    if (lane == 0) {
      logits[(size_t)e * H_ + h] = s;
      atomicMax(&mbuf[dst * H_ + h], fenc(s));     // segment max
    }
  }
}

__global__ void kgat_exp(const float* __restrict__ logits, const unsigned* __restrict__ mbuf,
                         const int* __restrict__ eidx, float* __restrict__ exb,
                         float* __restrict__ denom)
{
  int t = blockIdx.x * 256 + threadIdx.x;
  if (t >= ET_ * H_) return;
  int e = t >> 2, h = t & 3;
  int dst = (e < E_) ? eidx[E_ + e] : (e - E_);
  float m  = fdec(mbuf[dst * H_ + h]);
  float ex = expf(logits[t] - m);
  exb[t] = ex;
  atomicAdd(&denom[dst * H_ + h], ex);             // segment sum
}

__global__ void kgat_agg(const float* __restrict__ xl, const float* __restrict__ exb,
                         const float* __restrict__ denom, const int* __restrict__ eidx,
                         float* __restrict__ acc)
{
  int wave = threadIdx.x >> 5, lane = threadIdx.x & 31;
  int e = blockIdx.x * 8 + wave;
  if (e >= ET_) return;
  int src = (e < E_) ? eidx[e]      : (e - E_);
  int dst = (e < E_) ? eidx[E_ + e] : (e - E_);
  float al[H_];
  #pragma unroll
  for (int h = 0; h < H_; ++h) al[h] = exb[(size_t)e * H_ + h] / denom[dst * H_ + h];
  const float* pl = xl + (size_t)src * (H_ * D_);
  float* pa = acc + (size_t)dst * (H_ * D_);
  #pragma unroll
  for (int h = 0; h < H_; ++h) {
    #pragma unroll
    for (int j = 0; j < 8; ++j) {
      int idx = h * D_ + lane + j * 32;
      atomicAdd(&pa[idx], al[h] * pl[idx]);
    }
  }
}

// mean over heads + gb, then leaky 0.01
__global__ void kgat_post(const float* __restrict__ acc, const float* __restrict__ gb,
                          float* __restrict__ gpre)
{
  int t = blockIdx.x * 256 + threadIdx.x;
  if (t >= NT_ * D_) return;
  int n = t >> 8, c = t & 255;
  const float* p = acc + (size_t)n * (H_ * D_);
  float s = (p[c] + p[D_ + c] + p[2 * D_ + c] + p[3 * D_ + c]) * 0.25f + gb[c];
  gpre[t] = s > 0.f ? s : 0.01f * s;
}

// ---------------------------------------------------------------------------
// Final cross-attention: wave per (b,h,q); keys 15..29 valid; hd=64
// ---------------------------------------------------------------------------
__global__ void kattn(const float* __restrict__ Qb, const float* __restrict__ Kb,
                      const float* __restrict__ Vb, float* __restrict__ apre)
{
  __shared__ float als[H_][32];
  int h = threadIdx.x >> 5, lane = threadIdx.x & 31;
  int bq = blockIdx.x;
  int b = bq / NVH_, qi = bq % NVH_;
  const float* qp = Qb + (size_t)(b * NVH_ + qi) * D_ + h * 64;
  float s = -INFINITY;
  int k = lane;
  if (k >= NVH_ && k < NN_) {
    const float* kp = Kb + (size_t)(b * NN_ + k) * D_ + h * 64;
    float a = 0.f;
    #pragma unroll
    for (int d = 0; d < 64; ++d) a += qp[d] * kp[d];
    s = a * 0.125f;                               // / sqrt(64)
  }
  float m = wave_max(s);
  float e = (k >= NVH_ && k < NN_) ? expf(s - m) : 0.f;
  float sum = wave_sum(e);
  als[h][lane] = e / sum;
  __syncthreads();
  float* ap = apre + (size_t)(b * NVH_ + qi) * D_ + h * 64;
  for (int d = lane; d < 64; d += 32) {
    float a = 0.f;
    for (int k2 = NVH_; k2 < NN_; ++k2)
      a += als[h][k2] * Vb[(size_t)(b * NN_ + k2) * D_ + h * 64 + d];
    ap[d] = a;
  }
}

// gather q rows (first 15 nodes of each batch) from f16 buffer
__global__ void kbuildq(const _Float16* __restrict__ v16, _Float16* __restrict__ q16) {
  int t = blockIdx.x * 256 + threadIdx.x;
  if (t >= B_ * NVH_ * D_) return;
  int r = t >> 8, c = t & 255;
  int b = r / NVH_, qi = r % NVH_;
  q16[t] = v16[(size_t)(b * NN_ + qi) * D_ + c];
}
// hvh = h[:, :15, :]
__global__ void kcopy_hvh(const float* __restrict__ h, float* __restrict__ hvh) {
  int t = blockIdx.x * 256 + threadIdx.x;
  if (t >= B_ * NVH_ * D_) return;
  int r = t >> 8, c = t & 255;
  int b = r / NVH_, qi = r % NVH_;
  hvh[t] = h[(size_t)(b * NN_ + qi) * D_ + c];
}
// ef16[row, 0:256]=hvh[b,src], [256:512]=hvh[b,dst]; pairs (s,d), s-major, d skips s
__global__ void kef(const float* __restrict__ hvh, _Float16* __restrict__ ef) {
  int t = blockIdx.x * 256 + threadIdx.x;
  if (t >= EROWS_ * 2 * D_) return;
  int row = t >> 9, col = t & 511;
  int b = row / PAIRS_, p = row % PAIRS_;
  int s = p / 14, rr = p % 14;
  int d = rr + (rr >= s ? 1 : 0);
  int node = (col < D_) ? s : d;
  int c = col & 255;
  ef[t] = (_Float16)hvh[(size_t)(b * NVH_ + node) * D_ + c];
}
// final dot: out[row] = o3[row,:]·m4W + m4b
__global__ void km4(const float* __restrict__ o3, const float* __restrict__ w,
                    const float* __restrict__ bias, float* __restrict__ out)
{
  int wave = threadIdx.x >> 5, lane = threadIdx.x & 31;
  int row = blockIdx.x * 8 + wave;
  if (row >= EROWS_) return;
  float s = 0.f;
  #pragma unroll
  for (int j = 0; j < 4; ++j) {
    int c = lane + j * 32;
    s += o3[(size_t)row * 128 + c] * w[c];
  }
  s = wave_sum(s);
  if (lane == 0) out[row] = s + bias[0];
}

// ---------------------------------------------------------------------------
// Host side
// ---------------------------------------------------------------------------
static inline char* bump(char*& p, size_t bytes) {
  char* r = p;
  p += (bytes + 255) & ~size_t(255);
  return r;
}
static inline void gemm(hipStream_t s, const _Float16* A, const _Float16* BT,
                        const float* bias, float* C, _Float16* C16,
                        int M, int N, int K, int act, int acc) {
  int ntiles = (M >> 4) * ((N >> 4) >> 2);     // 16x64 strip per wave
  kgemm<<<CDIV(ntiles, 8), 256, 0, s>>>(A, BT, bias, C, C16, M, N, K, act, acc);
}
static inline void cvt(hipStream_t s, const float* src, _Float16* dst, int n) {
  kcvt<<<CDIV(n, 256), 256, 0, s>>>(src, dst, n);
}
static inline void cvtT(hipStream_t s, const float* src, _Float16* dst, int K, int N) {
  kcvtT<<<CDIV(K * N, 256), 256, 0, s>>>(src, dst, K, N);
}
static inline void zero(hipStream_t s, float* p, int n) {
  kzero<<<CDIV(n, 256), 256, 0, s>>>(p, n);
}
static inline void ln(hipStream_t s, const float* X, const float* g, const float* b,
                      float* Y, _Float16* Y16, int rows) {
  kln<<<CDIV(rows, 8), 256, 0, s>>>(X, g, b, Y, Y16, rows);
}

extern "C" void kernel_launch(void* const* d_in, const int* in_sizes, int n_in,
                              void* d_out, int out_size, void* d_ws, size_t ws_size,
                              hipStream_t stream)
{
  (void)in_sizes; (void)n_in; (void)out_size; (void)ws_size;

  // ---- input mapping: setup_inputs() dict insertion order, recursively ----
  // 0: x ; per layer l (base = 1 + 16*l):
  //   +0 Wl +1 bl +2 Wr +3 br +4 att +5 gb +6 gat_g +7 gat_b
  //   +8 cin_W +9 cin_b +10 conv_W +11 conv_b +12 cnn_g +13 cnn_b +14 fuse_g +15 fuse_b
  // 49..76: kW kb k_g k_b v_g v_b qW qb kkW kkb vW vb oW ob ffn_g ffn_b
  //         f1W f1b f2W f2b m1W m1b m2W m2b m3W m3b m4W m4b ; 77: edge_index
  const float* x = (const float*)d_in[0];
  auto fp = [&](int i) { return (const float*)d_in[i]; };
  const int* eidx = (const int*)d_in[77];

  const float *kW = fp(49), *kb = fp(50), *k_g = fp(51), *k_b = fp(52);
  const float *v_g = fp(53), *v_b = fp(54), *qW = fp(55), *qb = fp(56);
  const float *kkW = fp(57), *kkb = fp(58), *vW = fp(59), *vb = fp(60);
  const float *oW = fp(61), *ob = fp(62), *ffn_g = fp(63), *ffn_b = fp(64);
  const float *f1W = fp(65), *f1b = fp(66), *f2W = fp(67), *f2b = fp(68);
  const float *m1W = fp(69), *m1b = fp(70), *m2W = fp(71), *m2b = fp(72);
  const float *m3W = fp(73), *m3b = fp(74), *m4W = fp(75), *m4b = fp(76);

  // ---- workspace layout ----
  char* wp = (char*)d_ws;
  _Float16* h16     = (_Float16*)bump(wp, (size_t)NT_ * D_ * 2);
  _Float16* wl16    = (_Float16*)bump(wp, (size_t)D_ * H_ * D_ * 2);   // BT layouts
  _Float16* wr16    = (_Float16*)bump(wp, (size_t)D_ * H_ * D_ * 2);
  _Float16* cin16   = (_Float16*)bump(wp, (size_t)D_ * D_ * 2);
  _Float16* convw16 = (_Float16*)bump(wp, (size_t)7 * D_ * D_ * 2);
  float*    xl      = (float*)bump(wp, (size_t)NT_ * H_ * D_ * 4);
  float*    xr      = (float*)bump(wp, (size_t)NT_ * H_ * D_ * 4);
  float*    logits  = (float*)bump(wp, (size_t)ET_ * H_ * 4);
  float*    exb     = (float*)bump(wp, (size_t)ET_ * H_ * 4);
  float*    mbuf    = (float*)bump(wp, (size_t)NT_ * H_ * 4);   // as unsigned
  float*    denom   = (float*)bump(wp, (size_t)NT_ * H_ * 4);
  float*    gacc    = (float*)bump(wp, (size_t)NT_ * H_ * D_ * 4);
  float*    gpre    = (float*)bump(wp, (size_t)NT_ * D_ * 4);
  float*    gbuf    = (float*)bump(wp, (size_t)NT_ * D_ * 4);
  float*    xc      = (float*)bump(wp, (size_t)NT_ * D_ * 4);
  _Float16* pad16   = (_Float16*)bump(wp, (size_t)B_ * 36 * D_ * 2);
  float*    convout = (float*)bump(wp, (size_t)NT_ * D_ * 4);
  float*    cbuf    = (float*)bump(wp, (size_t)NT_ * D_ * 4);
  float*    hA      = (float*)bump(wp, (size_t)NT_ * D_ * 4);
  float*    hB      = (float*)bump(wp, (size_t)NT_ * D_ * 4);
  // final stage
  float*    kpre    = (float*)bump(wp, (size_t)NT_ * D_ * 4);
  float*    kctx    = (float*)bump(wp, (size_t)NT_ * D_ * 4);
  _Float16* kctx16  = (_Float16*)bump(wp, (size_t)NT_ * D_ * 2);
  float*    vctx    = (float*)bump(wp, (size_t)NT_ * D_ * 4);
  _Float16* vctx16  = (_Float16*)bump(wp, (size_t)NT_ * D_ * 2);
  _Float16* q16     = (_Float16*)bump(wp, (size_t)B_ * NVH_ * D_ * 2);
  float*    Qb      = (float*)bump(wp, (size_t)B_ * NVH_ * D_ * 4);
  float*    Kb      = (float*)bump(wp, (size_t)NT_ * D_ * 4);
  float*    Vb      = (float*)bump(wp, (size_t)NT_ * D_ * 4);
  float*    apre    = (float*)bump(wp, (size_t)B_ * NVH_ * D_ * 4);
  _Float16* apre16  = (_Float16*)bump(wp, (size_t)B_ * NVH_ * D_ * 2);
  float*    hvh     = (float*)bump(wp, (size_t)B_ * NVH_ * D_ * 4);
  float*    fln     = (float*)bump(wp, (size_t)B_ * NVH_ * D_ * 4);
  _Float16* fln16   = (_Float16*)bump(wp, (size_t)B_ * NVH_ * D_ * 2);
  float*    f1out   = (float*)bump(wp, (size_t)B_ * NVH_ * 4 * D_ * 4);
  _Float16* f1out16 = (_Float16*)bump(wp, (size_t)B_ * NVH_ * 4 * D_ * 2);
  _Float16* ef16    = (_Float16*)bump(wp, (size_t)EROWS_ * 2 * D_ * 2);
  float*    o1      = (float*)bump(wp, (size_t)EROWS_ * 256 * 4);
  _Float16* o116    = (_Float16*)bump(wp, (size_t)EROWS_ * 256 * 2);
  float*    o2      = (float*)bump(wp, (size_t)EROWS_ * 128 * 4);
  _Float16* o216    = (_Float16*)bump(wp, (size_t)EROWS_ * 128 * 2);
  float*    o3      = (float*)bump(wp, (size_t)EROWS_ * 128 * 4);
  _Float16* kW16    = (_Float16*)bump(wp, (size_t)D_ * D_ * 2);
  _Float16* qW16    = (_Float16*)bump(wp, (size_t)D_ * D_ * 2);
  _Float16* kkW16   = (_Float16*)bump(wp, (size_t)D_ * D_ * 2);
  _Float16* vW16    = (_Float16*)bump(wp, (size_t)D_ * D_ * 2);
  _Float16* oW16    = (_Float16*)bump(wp, (size_t)D_ * D_ * 2);
  _Float16* f1W16   = (_Float16*)bump(wp, (size_t)D_ * 4 * D_ * 2);
  _Float16* f2W16   = (_Float16*)bump(wp, (size_t)4 * D_ * D_ * 2);
  _Float16* m1W16   = (_Float16*)bump(wp, (size_t)2 * D_ * 256 * 2);
  _Float16* m2W16   = (_Float16*)bump(wp, (size_t)256 * 128 * 2);
  _Float16* m3W16   = (_Float16*)bump(wp, (size_t)128 * 128 * 2);

  // ---- 3 GNN layers ----
  const float* hcur = x;
  float* hout[3] = { hA, hB, hA };
  for (int l = 0; l < 3; ++l) {
    int base = 1 + 16 * l;
    const float *Wl = fp(base + 0), *bl = fp(base + 1), *Wr = fp(base + 2), *br = fp(base + 3);
    const float *att = fp(base + 4), *gb = fp(base + 5), *gat_g = fp(base + 6), *gat_b = fp(base + 7);
    const float *cinW = fp(base + 8), *cin_b = fp(base + 9), *convW = fp(base + 10), *conv_b = fp(base + 11);
    const float *cnn_g = fp(base + 12), *cnn_b = fp(base + 13), *fuse_g = fp(base + 14), *fuse_b = fp(base + 15);

    cvtT(stream, Wl, wl16, D_, H_ * D_);
    cvtT(stream, Wr, wr16, D_, H_ * D_);
    cvtT(stream, cinW, cin16, D_, D_);
    kcvtconv<<<CDIV(7 * D_ * D_, 256), 256, 0, stream>>>(convW, convw16);
    cvt(stream, hcur, h16, NT_ * D_);

    gemm(stream, h16, wl16, bl, xl, nullptr, NT_, H_ * D_, D_, ACT_NONE, 0);
    gemm(stream, h16, wr16, br, xr, nullptr, NT_, H_ * D_, D_, ACT_NONE, 0);

    zero(stream, mbuf, NT_ * H_);
    zero(stream, denom, NT_ * H_);
    zero(stream, gacc, NT_ * H_ * D_);
    kgat_logits<<<CDIV(ET_, 8), 256, 0, stream>>>(xl, xr, att, eidx, logits, (unsigned*)mbuf);
    kgat_exp<<<CDIV(ET_ * H_, 256), 256, 0, stream>>>(logits, (unsigned*)mbuf, eidx, exb, denom);
    kgat_agg<<<CDIV(ET_, 8), 256, 0, stream>>>(xl, exb, denom, eidx, gacc);
    kgat_post<<<CDIV(NT_ * D_, 256), 256, 0, stream>>>(gacc, gb, gpre);
    ln(stream, gpre, gat_g, gat_b, gbuf, nullptr, NT_);

    gemm(stream, h16, cin16, cin_b, xc, nullptr, NT_, D_, D_, ACT_NONE, 0);
    kpad<<<CDIV(B_ * 36 * D_, 256), 256, 0, stream>>>(xc, pad16);
    kconvgemm<<<CDIV((NT_ / 16) * (D_ / 64), 8), 256, 0, stream>>>(pad16, convw16, conv_b, convout);
    ln(stream, convout, cnn_g, cnn_b, cbuf, nullptr, NT_);

    kadd3<<<CDIV(NT_ * D_, 256), 256, 0, stream>>>(gbuf, cbuf, hcur, NT_ * D_);
    ln(stream, gbuf, fuse_g, fuse_b, hout[l], nullptr, NT_);
    hcur = hout[l];
  }

  // ---- final stage ----
  cvt(stream, x, h16, NT_ * D_);                      // reuse h16 as x16
  cvtT(stream, kW, kW16, D_, D_);
  gemm(stream, h16, kW16, kb, kpre, nullptr, NT_, D_, D_, ACT_NONE, 0);
  ln(stream, kpre, k_g, k_b, kctx, kctx16, NT_);
  ln(stream, hcur, v_g, v_b, vctx, vctx16, NT_);

  kbuildq<<<CDIV(B_ * NVH_ * D_, 256), 256, 0, stream>>>(vctx16, q16);
  cvtT(stream, qW, qW16, D_, D_);
  cvtT(stream, kkW, kkW16, D_, D_);
  cvtT(stream, vW, vW16, D_, D_);
  cvtT(stream, oW, oW16, D_, D_);
  gemm(stream, q16, qW16, qb, Qb, nullptr, B_ * NVH_, D_, D_, ACT_NONE, 0);
  gemm(stream, kctx16, kkW16, kkb, Kb, nullptr, NT_, D_, D_, ACT_NONE, 0);
  gemm(stream, vctx16, vW16, vb, Vb, nullptr, NT_, D_, D_, ACT_NONE, 0);

  kattn<<<B_ * NVH_, 128, 0, stream>>>(Qb, Kb, Vb, apre);
  cvt(stream, apre, apre16, B_ * NVH_ * D_);
  kcopy_hvh<<<CDIV(B_ * NVH_ * D_, 256), 256, 0, stream>>>(hcur, hvh);
  gemm(stream, apre16, oW16, ob, hvh, nullptr, B_ * NVH_, D_, D_, ACT_NONE, 1); // hvh += a_out

  ln(stream, hvh, ffn_g, ffn_b, fln, fln16, B_ * NVH_);
  cvtT(stream, f1W, f1W16, D_, 4 * D_);
  cvtT(stream, f2W, f2W16, 4 * D_, D_);
  gemm(stream, fln16, f1W16, f1b, f1out, f1out16, B_ * NVH_, 4 * D_, D_, ACT_GELU, 0);
  gemm(stream, f1out16, f2W16, f2b, hvh, nullptr, B_ * NVH_, D_, 4 * D_, ACT_NONE, 1); // residual

  kef<<<CDIV(EROWS_ * 2 * D_, 256), 256, 0, stream>>>(hvh, ef16);
  cvtT(stream, m1W, m1W16, 2 * D_, 256);
  cvtT(stream, m2W, m2W16, 256, 128);
  cvtT(stream, m3W, m3W16, 128, 128);
  gemm(stream, ef16, m1W16, m1b, o1, o116, EROWS_, 256, 2 * D_, ACT_LEAKY, 0);
  gemm(stream, o116, m2W16, m2b, o2, o216, EROWS_, 128, 256, ACT_LEAKY, 0);
  gemm(stream, o216, m3W16, m3b, o3, nullptr, EROWS_, 128, 128, ACT_LEAKY, 0);
  km4<<<CDIV(EROWS_, 8), 256, 0, stream>>>(o3, m4W, m4b, (float*)d_out);
}
